// SoftNeuralDictionary_42399917146657
// MI455X (gfx1250) — compile-verified, matrix operand order checked by reading, not run on hardware
//
#include <hip/hip_runtime.h>
#include <hip/hip_bf16.h>

// ---------------------------------------------------------------------------
// Soft neural dictionary  (flash-decoding fused kernel, CDNA5 / gfx1250)
//   scores = (x/temp) . keys^T   -> online softmax ->  out = P . values
//
// Grid = (query_tiles, NSPLIT): each workgroup sweeps one capacity slice for
// one 16-query tile (8 wave32 waves, bf16 WMMA both GEMMs, f32 accum), writing
// either the final normalized output (NSPLIT==1) or an unnormalized partial
// accumulator + per-row (m,l) stats that a combine kernel merges.
// NSPLIT is chosen on the host as the largest split whose partials fit d_ws.
// ---------------------------------------------------------------------------

typedef __attribute__((ext_vector_type(16))) __bf16 bf16x16;
typedef __attribute__((ext_vector_type(8)))  __bf16 bf16x8;
typedef __attribute__((ext_vector_type(4)))  __bf16 bf16x4;
typedef __attribute__((ext_vector_type(4)))  float  f32x4;
typedef __attribute__((ext_vector_type(8)))  float  v8f;

union Frag16 { bf16x16 v; bf16x8 h[2]; };

static constexpr int KD     = 512;       // key dim
static constexpr int VD     = 512;       // value dim
static constexpr int CAP    = 65536;     // capacity
static constexpr int MT     = 16;        // queries per workgroup
static constexpr int CHUNK  = 128;       // capacity keys per workgroup iteration
static constexpr int NWAVE  = 8;
static constexpr int PITCHX = KD + 8;    // bf16 elems; row stride 1040B (16B mult)
static constexpr int PITCHC = CHUNK + 8; // bf16 elems; row stride 272B  (16B mult)

// A/B 16-bit WMMA fragment (16x32 / 32x16): lane holds two contiguous runs of 8
// 16-bit elements:  K = half*8 + {0..7}  and  K = 16 + half*8 + {0..7}.
__device__ __forceinline__ bf16x16 frag_lds(const __bf16* row, int k0, int half) {
    Frag16 f;
    f.h[0] = *(const bf16x8*)(row + k0 + half * 8);
    f.h[1] = *(const bf16x8*)(row + k0 + 16 + half * 8);
    return f.v;
}

// Same fragment sourced from an f32 row in global memory (convert to bf16).
__device__ __forceinline__ bf16x16 frag_global_f32(const float* row, int k0, int half) {
    Frag16 f;
    const float* p0 = row + k0 + half * 8;
    const float* p1 = row + k0 + 16 + half * 8;
    f32x4 a0 = *(const f32x4*)(p0);
    f32x4 a1 = *(const f32x4*)(p0 + 4);
    f32x4 b0 = *(const f32x4*)(p1);
    f32x4 b1 = *(const f32x4*)(p1 + 4);
#pragma unroll
    for (int j = 0; j < 4; ++j) {
        f.h[0][j]     = (__bf16)a0[j];
        f.h[0][j + 4] = (__bf16)a1[j];
        f.h[1][j]     = (__bf16)b0[j];
        f.h[1][j + 4] = (__bf16)b1[j];
    }
    return f.v;
}

__device__ __forceinline__ float redmax16(float v) {
    v = fmaxf(v, __shfl_xor(v, 1));
    v = fmaxf(v, __shfl_xor(v, 2));
    v = fmaxf(v, __shfl_xor(v, 4));
    v = fmaxf(v, __shfl_xor(v, 8));
    return v;   // uniform within each 16-lane half
}
__device__ __forceinline__ float redsum16(float v) {
    v += __shfl_xor(v, 1);
    v += __shfl_xor(v, 2);
    v += __shfl_xor(v, 4);
    v += __shfl_xor(v, 8);
    return v;
}

__global__ __launch_bounds__(256)
void snd_flash_kernel(const float* __restrict__ x,
                      const float* __restrict__ keys,
                      const float* __restrict__ values,
                      const float* __restrict__ log_temp,
                      float* __restrict__ out,        // final output (nsplit==1)
                      float* __restrict__ pacc,       // partial acc   (nsplit>1)
                      float* __restrict__ pstats,     // per-row (m,l) (nsplit>1)
                      int cnum, int batch) {
    // LDS budget: 16*520*2 + 16*136*2 + 512*136*2 + 2*8*16*4  ~= 158 KB (<320KB/WGP)
    __shared__ __bf16 xs[MT * PITCHX];        // x tile (pre-scaled, bf16)
    __shared__ __bf16 ps[MT * PITCHC];        // softmax probabilities (bf16)
    __shared__ __bf16 vt[VD * PITCHC];        // values chunk, transposed: vt[d][c]
    __shared__ float  red_max[NWAVE][MT];
    __shared__ float  red_sum[NWAVE][MT];

    const int tid   = threadIdx.x;
    const int lane  = tid & 31;
    const int w     = tid >> 5;           // wave id 0..7
    const int ln    = lane & 15;
    const int half  = lane >> 4;          // 0: rows g, 1: rows g+8 (C/D layout)
    const int qbase = blockIdx.x * MT;
    const int split = blockIdx.y;
    const int c0all = split * cnum;       // this workgroup's capacity slice
    const int dbase = w * 64;             // this wave's 64 output columns

    const float inv_t = __expf(-log_temp[0]);

    // ---- stage x tile into LDS as bf16, folding the temperature scale -------
    {
        const int r   = tid >> 4;          // 0..15
        const int col = (tid & 15) * 32;   // 0..480
        const float* xr = x + (size_t)(qbase + r) * KD + col;
        __bf16* dst = xs + r * PITCHX + col;
#pragma unroll
        for (int j = 0; j < 4; ++j) {
            f32x4 u = *(const f32x4*)(xr + j * 8);
            f32x4 v = *(const f32x4*)(xr + j * 8 + 4);
            bf16x8 hvec;
#pragma unroll
            for (int e = 0; e < 4; ++e) {
                hvec[e]     = (__bf16)(u[e] * inv_t);
                hvec[e + 4] = (__bf16)(v[e] * inv_t);
            }
            *(bf16x8*)(dst + j * 8) = hvec;
        }
    }

    // ---- online softmax state (row g + 8*half lives in element g) ----------
    float m_run[8], l_run[8];
    v8f acc[4];
#pragma unroll
    for (int g = 0; g < 8; ++g) { m_run[g] = -__builtin_inff(); l_run[g] = 0.0f; }
#pragma unroll
    for (int t = 0; t < 4; ++t)
#pragma unroll
        for (int g = 0; g < 8; ++g) acc[t][g] = 0.0f;

    const __bf16* xrow = xs + ln * PITCHX;   // A-fragment source row (query ln)

    const int niter = cnum / CHUNK;
    for (int it = 0; it < niter; ++it) {
        const int cbase = c0all + it * CHUNK;
        const int ck    = cbase + w * 16;    // this wave's 16 keys of the chunk

        // prefetch next chunk (keys row for this lane, values block head)
        if (it + 1 < niter) {
            __builtin_prefetch(keys + (size_t)(ck + CHUNK + ln) * KD, 0, 1);
            __builtin_prefetch(values + (size_t)(cbase + CHUNK) * VD + lane * 16, 0, 1);
        }

        // ---- scores: S(16x16) = Xtile(16x512) . Ktile(16x512)^T ------------
        v8f S;
#pragma unroll
        for (int g = 0; g < 8; ++g) S[g] = 0.0f;
        const float* krow = keys + (size_t)(ck + ln) * KD;
#pragma unroll 4
        for (int kk = 0; kk < KD / 32; ++kk) {
            const int k0 = kk * 32;
            bf16x16 a = frag_lds(xrow, k0, half);
            bf16x16 b = frag_global_f32(krow, k0, half);
            S = __builtin_amdgcn_wmma_f32_16x16x32_bf16(
                    false, a, false, b, (short)0, S, false, false);
        }

        __syncthreads();   // (A) previous iteration's reads of vt/ps/red done

        // ---- stage values chunk transposed into LDS (coalesced b128 reads) --
        {
            const int c0 = (tid & 31) * 4;        // 4 consecutive capacity rows
            const int dg = (tid >> 5) * 8;        // base of 8 value columns
#pragma unroll
            for (int rep = 0; rep < 8; ++rep) {
                const int d0 = dg + rep * 64;     // covers d = 0..511
                f32x4 lo[4], hi[4];
#pragma unroll
                for (int r = 0; r < 4; ++r) {
                    const float* vr = values + (size_t)(cbase + c0 + r) * VD + d0;
                    lo[r] = *(const f32x4*)(vr);
                    hi[r] = *(const f32x4*)(vr + 4);
                }
#pragma unroll
                for (int j = 0; j < 4; ++j) {
                    bf16x4 p, q;
#pragma unroll
                    for (int r = 0; r < 4; ++r) {
                        p[r] = (__bf16)lo[r][j];
                        q[r] = (__bf16)hi[r][j];
                    }
                    *(bf16x4*)&vt[(size_t)(d0 + j)     * PITCHC + c0] = p;
                    *(bf16x4*)&vt[(size_t)(d0 + 4 + j) * PITCHC + c0] = q;
                }
            }
        }

        // ---- per-wave row max -> LDS ---------------------------------------
        float rm[8];
#pragma unroll
        for (int g = 0; g < 8; ++g) rm[g] = redmax16(S[g]);
        if (ln == 0) {
#pragma unroll
            for (int g = 0; g < 8; ++g) red_max[w][g + 8 * half] = rm[g];
        }
        __syncthreads();   // (B)

        // ---- combine maxes, compute P = exp(S - m_new), partial sums -------
        float m_new[8], scale[8], rs[8];
#pragma unroll
        for (int g = 0; g < 8; ++g) {
            const int row = g + 8 * half;
            float cm = red_max[0][row];
#pragma unroll
            for (int w2 = 1; w2 < NWAVE; ++w2) cm = fmaxf(cm, red_max[w2][row]);
            m_new[g] = fmaxf(m_run[g], cm);
            scale[g] = __expf(m_run[g] - m_new[g]);   // 0 on first iteration
            m_run[g] = m_new[g];

            float p = __expf(S[g] - m_new[g]);
            ps[row * PITCHC + w * 16 + ln] = (__bf16)p;
            rs[g] = redsum16(p);
        }
        if (ln == 0) {
#pragma unroll
            for (int g = 0; g < 8; ++g) red_sum[w][g + 8 * half] = rs[g];
        }
        __syncthreads();   // (C) P, sums and vt visible to everyone

        // ---- update l, rescale accumulators --------------------------------
#pragma unroll
        for (int g = 0; g < 8; ++g) {
            const int row = g + 8 * half;
            float cs = 0.0f;
#pragma unroll
            for (int w2 = 0; w2 < NWAVE; ++w2) cs += red_sum[w2][row];
            l_run[g] = l_run[g] * scale[g] + cs;
        }
#pragma unroll
        for (int t = 0; t < 4; ++t)
#pragma unroll
            for (int g = 0; g < 8; ++g) acc[t][g] *= scale[g];

        // ---- out += P(16x128) . V(128x64-slice), all fragments from LDS ----
        const __bf16* prow = ps + ln * PITCHC;
#pragma unroll
        for (int kc = 0; kc < CHUNK / 32; ++kc) {
            bf16x16 pa = frag_lds(prow, kc * 32, half);
#pragma unroll
            for (int t = 0; t < 4; ++t) {
                const __bf16* vtrow = vt + (size_t)(dbase + t * 16 + ln) * PITCHC;
                bf16x16 vb = frag_lds(vtrow, kc * 32, half);
                acc[t] = __builtin_amdgcn_wmma_f32_16x16x32_bf16(
                             false, pa, false, vb, (short)0, acc[t], false, false);
            }
        }
        // loop-end barrier is barrier (A) of the next iteration
    }

    // ---- epilogue ----------------------------------------------------------
    if (pstats != nullptr) {
        // partial mode: unnormalized acc + per-row (m,l)
        if (w == 0 && ln == 0) {
#pragma unroll
            for (int g = 0; g < 8; ++g) {
                const size_t row = (size_t)split * batch + qbase + g + 8 * half;
                pstats[row * 2 + 0] = m_run[g];
                pstats[row * 2 + 1] = l_run[g];
            }
        }
#pragma unroll
        for (int t = 0; t < 4; ++t)
#pragma unroll
            for (int g = 0; g < 8; ++g) {
                const size_t row = (size_t)split * batch + qbase + g + 8 * half;
                pacc[row * VD + dbase + t * 16 + ln] = acc[t][g];
            }
    } else {
        // single-pass mode: normalize and store final output
#pragma unroll
        for (int t = 0; t < 4; ++t)
#pragma unroll
            for (int g = 0; g < 8; ++g) {
                const int row = qbase + g + 8 * half;
                const int col = dbase + t * 16 + ln;
                out[(size_t)row * VD + col] = acc[t][g] / l_run[g];
            }
    }
}

// ---- combine kernel: merge NSPLIT partials with the softmax merge rule -----
__global__ __launch_bounds__(256)
void snd_combine_kernel(const float* __restrict__ pacc,
                        const float* __restrict__ pstats,
                        float* __restrict__ out,
                        int nsplit, int batch) {
    const int row = blockIdx.x;
    float sc[8];
    float M = -__builtin_inff();
    for (int s = 0; s < nsplit; ++s)
        M = fmaxf(M, pstats[((size_t)s * batch + row) * 2 + 0]);
    float L = 0.0f;
    for (int s = 0; s < nsplit; ++s) {
        const float ms = pstats[((size_t)s * batch + row) * 2 + 0];
        const float ls = pstats[((size_t)s * batch + row) * 2 + 1];
        sc[s] = __expf(ms - M);
        L += ls * sc[s];
    }
    const float invL = 1.0f / L;
    for (int col = threadIdx.x; col < VD; col += blockDim.x) {
        float a = 0.0f;
        for (int s = 0; s < nsplit; ++s)
            a += pacc[((size_t)s * batch + row) * VD + col] * sc[s];
        out[(size_t)row * VD + col] = a * invL;
    }
}

extern "C" void kernel_launch(void* const* d_in, const int* in_sizes, int n_in,
                              void* d_out, int out_size, void* d_ws, size_t ws_size,
                              hipStream_t stream) {
    (void)n_in; (void)out_size;
    const float* x        = (const float*)d_in[0];
    const float* keys     = (const float*)d_in[1];
    const float* values   = (const float*)d_in[2];
    const float* log_temp = (const float*)d_in[3];
    float* out            = (float*)d_out;

    const int batch    = in_sizes[0] / KD;        // 1024
    const int blocks_m = batch / MT;              // 64 query tiles

    // largest capacity split whose partial buffers fit in the workspace
    int nsplit = 1;
    for (int s = 8; s > 1; s >>= 1) {
        const size_t need = (size_t)s * batch * VD * sizeof(float)     // pacc
                          + (size_t)s * batch * 2  * sizeof(float);    // pstats
        if (ws_size >= need) { nsplit = s; break; }
    }

    if (nsplit > 1) {
        float* pacc   = (float*)d_ws;
        float* pstats = pacc + (size_t)nsplit * batch * VD;
        dim3 grid(blocks_m, nsplit);
        snd_flash_kernel<<<grid, 256, 0, stream>>>(
            x, keys, values, log_temp, nullptr, pacc, pstats, CAP / nsplit, batch);
        snd_combine_kernel<<<batch, 256, 0, stream>>>(pacc, pstats, out, nsplit, batch);
    } else {
        dim3 grid(blocks_m, 1);
        snd_flash_kernel<<<grid, 256, 0, stream>>>(
            x, keys, values, log_temp, out, nullptr, nullptr, CAP, batch);
    }
}
